// MoeRegLayer_16922171146616
// MI455X (gfx1250) — matrix operand order repaired and checked by reading, not compiled, not run on hardware
//
#include <hip/hip_runtime.h>
#include <hip/hip_bf16.h>
#include <math.h>

// ---------------------------------------------------------------------------
// MoE top-2 FFN for MI455X (gfx1250, wave32, WMMA bf16 16x16x32).
// Routed (top-2 only) => ~206 GFLOP instead of 824 GFLOP dense.
// fp32 weights streamed from HBM (~400MB mandatory), converted to bf16 in
// registers (v_cvt_pk_bf16_f32), staged in LDS, consumed by
// v_wmma_f32_16x16x32_bf16. bf16 A-tiles (x / h) are moved with the CDNA5
// async DMA path: global_load_async_to_lds_b128 + s_wait_asynccnt.
// Workspace need: ~81 MB (tokmap/cw/xbf16/h-buffer).
// ---------------------------------------------------------------------------

#define DIM   1024
#define HID   4096
#define NEXP  8
#define TM    128     // block tile rows (tokens)
#define TN    64      // block tile cols
#define TK    32      // k-step (WMMA K)
#define SA    40      // LDS row stride in bf16 elems (32 + 8 pad -> 80B, 16B aligned)
#define SB    40

typedef __bf16 bf16;
typedef __attribute__((ext_vector_type(8)))  bf16  bf16x8;
typedef __attribute__((ext_vector_type(16))) bf16  bf16x16;
typedef __attribute__((ext_vector_type(8)))  float f32x8;

__device__ __forceinline__ bf16x16 frag_cat(bf16x8 lo, bf16x8 hi) {
  return __builtin_shufflevector(lo, hi, 0,1,2,3,4,5,6,7,8,9,10,11,12,13,14,15);
}

// A fragment (16x32 bf16): lane holds row = lane&15, two contiguous 8-elem
// K-chunks at K = (lane<16 ? 0 : 8) and +16   (ISA 7.12.2 16-bit A layout)
__device__ __forceinline__ bf16x16 load_a(const bf16* As, int m0, int lane) {
  int r  = m0 + (lane & 15);
  int c0 = (lane & 16) ? 8 : 0;
  const bf16* p = As + r * SA + c0;
  bf16x8 lo = *(const bf16x8*)p;
  bf16x8 hi = *(const bf16x8*)(p + 16);
  return frag_cat(lo, hi);
}

// B fragment (32x16 bf16, column-major source = weight row contiguous in K):
// lane holds col = lane&15, two contiguous 8-elem K-chunks at K = (lane<16 ? 0 : 8) and +16
__device__ __forceinline__ bf16x16 load_b(const bf16* Bs, int n0, int lane) {
  int r  = n0 + (lane & 15);
  int c0 = (lane & 16) ? 8 : 0;
  const bf16* p = Bs + r * SB + c0;
  bf16x8 lo = *(const bf16x8*)p;
  bf16x8 hi = *(const bf16x8*)(p + 16);
  return frag_cat(lo, hi);
}

__device__ __forceinline__ void cvt8_store(bf16* d, float4 f0, float4 f1) {
  bf16x8 v;
  v[0] = (bf16)f0.x; v[1] = (bf16)f0.y; v[2] = (bf16)f0.z; v[3] = (bf16)f0.w;
  v[4] = (bf16)f1.x; v[5] = (bf16)f1.y; v[6] = (bf16)f1.z; v[7] = (bf16)f1.w;
  *(bf16x8*)d = v;
}

// Async DMA: copy 32 bytes global->LDS per lane (two b128s; the immediate
// offset is added to BOTH the LDS and global addresses per ISA 10.x).
__device__ __forceinline__ void async_copy32(unsigned lds_addr, const bf16* g) {
  asm volatile("global_load_async_to_lds_b128 %0, %1, off"
               :: "v"(lds_addr), "v"(g) : "memory");
  asm volatile("global_load_async_to_lds_b128 %0, %1, off offset:16"
               :: "v"(lds_addr), "v"(g) : "memory");
}
__device__ __forceinline__ void wait_async0() {
  asm volatile("s_wait_asynccnt 0x0" ::: "memory");
}

// ---------------------------------------------------------------------------
// 1) init: zero output, counters, tokmap = -1
__global__ void moe_init(float* out, int out_n, int* counts, int* tokmap, int rcap) {
  int i = blockIdx.x * blockDim.x + threadIdx.x;
  if (i < out_n) out[i] = 0.0f;
  if (i < rcap)  tokmap[i] = -1;
  if (i < NEXP)  counts[i] = 0;
}

// ---------------------------------------------------------------------------
// 2) gating: one wave per token; logits -> top-2 -> softmax -> counts
__global__ __launch_bounds__(256)
void moe_gate(const float* __restrict__ x, const float* __restrict__ gate_w,
              int* __restrict__ counts, int* __restrict__ top2e,
              float* __restrict__ top2p, int N) {
  __shared__ float gw[NEXP * DIM];                 // 32 KB
  int tid = threadIdx.x;
  for (int i = tid; i < NEXP * DIM; i += 256) gw[i] = gate_w[i];
  __syncthreads();
  int wave = tid >> 5, lane = tid & 31;
  int t = blockIdx.x * 8 + wave;
  if (t >= N) return;
  float acc[NEXP] = {};
  const float4* xp = (const float4*)(x + (size_t)t * DIM);
#pragma unroll
  for (int i = 0; i < DIM / 128; ++i) {
    float4 xv = xp[lane + i * 32];
#pragma unroll
    for (int e = 0; e < NEXP; ++e) {
      float4 gv = ((const float4*)(gw + e * DIM))[lane + i * 32];
      acc[e] += xv.x * gv.x + xv.y * gv.y + xv.z * gv.z + xv.w * gv.w;
    }
  }
#pragma unroll
  for (int e = 0; e < NEXP; ++e)
#pragma unroll
    for (int off = 16; off > 0; off >>= 1)
      acc[e] += __shfl_xor(acc[e], off, 32);
  if (lane == 0) {
    int e1 = 0; float v1 = acc[0];
    for (int e = 1; e < NEXP; ++e) if (acc[e] > v1) { v1 = acc[e]; e1 = e; }
    int e2 = (e1 == 0) ? 1 : 0; float v2 = acc[e2];
    for (int e = 0; e < NEXP; ++e)
      if (e != e1 && acc[e] > v2) { v2 = acc[e]; e2 = e; }
    float p1 = 1.0f / (1.0f + __expf(v2 - v1));   // softmax over top-2
    top2e[t * 2 + 0] = e1;  top2e[t * 2 + 1] = e2;
    top2p[t * 2 + 0] = p1;  top2p[t * 2 + 1] = 1.0f - p1;
    atomicAdd(counts + e1, 1);
    atomicAdd(counts + e2, 1);
  }
}

// ---------------------------------------------------------------------------
// 3) prefix: 128-padded per-expert segment bases
__global__ void moe_prefix(const int* counts, int* base, int* fill) {
  if (threadIdx.x == 0) {
    int acc = 0;
    for (int e = 0; e < NEXP; ++e) {
      base[e] = acc;
      acc += (counts[e] + TM - 1) & ~(TM - 1);
      fill[e] = 0;
    }
  }
}

// ---------------------------------------------------------------------------
// 4) scatter tokens into routed rows
__global__ void moe_scatter(const int* __restrict__ top2e, const float* __restrict__ top2p,
                            const int* __restrict__ base, int* __restrict__ fill,
                            int* __restrict__ tokmap, float* __restrict__ cw, int N) {
  int t = blockIdx.x * blockDim.x + threadIdx.x;
  if (t >= N) return;
#pragma unroll
  for (int j = 0; j < 2; ++j) {
    int e    = top2e[t * 2 + j];
    int slot = atomicAdd(fill + e, 1);
    int row  = base[e] + slot;
    tokmap[row] = t;
    cw[row]     = top2p[t * 2 + j];
  }
}

// ---------------------------------------------------------------------------
// 5) x -> bf16 once (A is reused across 64 N-tiles * 2 matrices)
__global__ void moe_cvt_x(const float* __restrict__ x, bf16* __restrict__ xbf, int total) {
  int i = (blockIdx.x * blockDim.x + threadIdx.x) * 8;
  if (i >= total) return;
  const float4* s = (const float4*)(x + i);
  cvt8_store(xbf + i, s[0], s[1]);
}

// ---------------------------------------------------------------------------
// 6) GEMM1 fused: g = x@wg^T, i = x@wi^T (shared A tile), h = silu(g+bg)*(i+bi)
__global__ __launch_bounds__(256)
void moe_ffn1(const bf16* __restrict__ xbf,
              const float* __restrict__ wg, const float* __restrict__ bg,
              const float* __restrict__ wi, const float* __restrict__ bi,
              const int* __restrict__ counts, const int* __restrict__ base,
              const int* __restrict__ tokmap, bf16* __restrict__ h, int tilesM) {
  int e  = blockIdx.y / tilesM;
  int mt = blockIdx.y % tilesM;
  int padded = (counts[e] + TM - 1) & ~(TM - 1);
  if (mt * TM >= padded) return;                    // block-uniform exit
  int rowbase = base[e] + mt * TM;
  int n0blk   = blockIdx.x * TN;

  __shared__ __align__(16) bf16 As[TM * SA];
  __shared__ __align__(16) bf16 Bg[TN * SB];
  __shared__ __align__(16) bf16 Bi[TN * SB];

  int tid = threadIdx.x, lane = tid & 31, wave = tid >> 5;
  int wm = wave >> 1, wn = wave & 1;                // 4x2 wave grid -> 128x64
  int arow = tid >> 1, ahalf = tid & 1;             // A staging: 128 rows x 2 halves
  int brow = tid >> 2, bq = tid & 3;                // B staging: 64 rows x 4 quads
  int tok  = tokmap[rowbase + arow];

  const float* wgp = wg + (size_t)e * HID * DIM + (size_t)(n0blk + brow) * DIM + bq * 8;
  const float* wip = wi + (size_t)e * HID * DIM + (size_t)(n0blk + brow) * DIM + bq * 8;
  const bf16*  xp  = xbf + (size_t)(tok >= 0 ? tok : 0) * DIM + ahalf * 16;
  unsigned ldsA = (unsigned)(uintptr_t)(As + arow * SA + ahalf * 16);

  // padded rows: their LDS slice is written only here; zero it once.
  if (tok < 0) {
    uint4 z = {0, 0, 0, 0};
    uint4* d = (uint4*)(As + arow * SA + ahalf * 16);
    d[0] = z; d[1] = z;
  }

  f32x8 ag[2][2] = {}, ai[2][2] = {};

  for (int k0 = 0; k0 < DIM; k0 += TK) {
    // A tile: bf16 already -> async DMA straight into LDS (no VGPR round-trip)
    if (tok >= 0) async_copy32(ldsA, xp + k0);
    // B tiles: fp32 -> bf16 in regs, then LDS
    { const float4* s = (const float4*)(wgp + k0); cvt8_store(Bg + brow * SB + bq * 8, s[0], s[1]); }
    { const float4* s = (const float4*)(wip + k0); cvt8_store(Bi + brow * SB + bq * 8, s[0], s[1]); }
    if (k0 + TK < DIM) {
      __builtin_prefetch(wgp + k0 + TK, 0, 1);      // global_prefetch_b8
      __builtin_prefetch(wip + k0 + TK, 0, 1);
    }
    wait_async0();                                  // own async LDS writes done
    __syncthreads();

    bf16x16 a0f = load_a(As, wm * 32 + 0,  lane);
    bf16x16 a1f = load_a(As, wm * 32 + 16, lane);
#pragma unroll
    for (int sn = 0; sn < 2; ++sn) {
      bf16x16 bgf = load_b(Bg, wn * 32 + sn * 16, lane);
      ag[0][sn] = __builtin_amdgcn_wmma_f32_16x16x32_bf16(false, a0f, false, bgf, (short)0, ag[0][sn], false, false);
      ag[1][sn] = __builtin_amdgcn_wmma_f32_16x16x32_bf16(false, a1f, false, bgf, (short)0, ag[1][sn], false, false);
      bf16x16 bif = load_b(Bi, wn * 32 + sn * 16, lane);
      ai[0][sn] = __builtin_amdgcn_wmma_f32_16x16x32_bf16(false, a0f, false, bif, (short)0, ai[0][sn], false, false);
      ai[1][sn] = __builtin_amdgcn_wmma_f32_16x16x32_bf16(false, a1f, false, bif, (short)0, ai[1][sn], false, false);
    }
    __syncthreads();
  }

  // epilogue: h = silu(g + bg) * (i + bi)  (C layout: VGPR r = row r / r+8)
#pragma unroll
  for (int sn = 0; sn < 2; ++sn) {
    int ncol = n0blk + wn * 32 + sn * 16 + (lane & 15);
    float bgv = bg[e * HID + ncol];
    float biv = bi[e * HID + ncol];
#pragma unroll
    for (int sm = 0; sm < 2; ++sm) {
      int mrow = rowbase + wm * 32 + sm * 16 + ((lane & 16) ? 8 : 0);
#pragma unroll
      for (int r = 0; r < 8; ++r) {
        float g  = ag[sm][sn][r] + bgv;
        float v  = ai[sm][sn][r] + biv;
        float hv = (g / (1.0f + __expf(-g))) * v;
        h[(size_t)(mrow + r) * HID + ncol] = (bf16)hv;
      }
    }
  }
}

// ---------------------------------------------------------------------------
// 7) GEMM2: y = h @ wo^T + bo; out[tok] += cw * y  (scatter via f32 atomics)
__global__ __launch_bounds__(256)
void moe_ffn2(const bf16* __restrict__ h,
              const float* __restrict__ wo, const float* __restrict__ bo,
              const int* __restrict__ counts, const int* __restrict__ base,
              const int* __restrict__ tokmap, const float* __restrict__ cw,
              float* __restrict__ out, int tilesM) {
  int e  = blockIdx.y / tilesM;
  int mt = blockIdx.y % tilesM;
  int padded = (counts[e] + TM - 1) & ~(TM - 1);
  if (mt * TM >= padded) return;
  int rowbase = base[e] + mt * TM;
  int n0blk   = blockIdx.x * TN;

  __shared__ __align__(16) bf16 As[TM * SA];
  __shared__ __align__(16) bf16 Bs[TN * SB];
  __shared__ int   tokS[TM];
  __shared__ float cwS[TM];

  int tid = threadIdx.x, lane = tid & 31, wave = tid >> 5;
  int wm = wave >> 1, wn = wave & 1;
  int arow = tid >> 1, ahalf = tid & 1;
  int brow = tid >> 2, bq = tid & 3;
  if (tid < TM) { tokS[tid] = tokmap[rowbase + tid]; cwS[tid] = cw[rowbase + tid]; }

  const bf16*  hp  = h  + (size_t)(rowbase + arow) * HID + ahalf * 16;
  const float* wop = wo + (size_t)e * DIM * HID + (size_t)(n0blk + brow) * HID + bq * 8;
  unsigned ldsA = (unsigned)(uintptr_t)(As + arow * SA + ahalf * 16);

  f32x8 acc[2][2] = {};
  for (int k0 = 0; k0 < HID; k0 += TK) {
    async_copy32(ldsA, hp + k0);                    // h is bf16: DMA to LDS
    { const float4* s = (const float4*)(wop + k0);
      cvt8_store(Bs + brow * SB + bq * 8, s[0], s[1]); }
    if (k0 + TK < HID) __builtin_prefetch(wop + k0 + TK, 0, 1);
    wait_async0();
    __syncthreads();

    bf16x16 a0f = load_a(As, wm * 32 + 0,  lane);
    bf16x16 a1f = load_a(As, wm * 32 + 16, lane);
#pragma unroll
    for (int sn = 0; sn < 2; ++sn) {
      bf16x16 bbf = load_b(Bs, wn * 32 + sn * 16, lane);
      acc[0][sn] = __builtin_amdgcn_wmma_f32_16x16x32_bf16(false, a0f, false, bbf, (short)0, acc[0][sn], false, false);
      acc[1][sn] = __builtin_amdgcn_wmma_f32_16x16x32_bf16(false, a1f, false, bbf, (short)0, acc[1][sn], false, false);
    }
    __syncthreads();
  }

#pragma unroll
  for (int sn = 0; sn < 2; ++sn) {
    int ncol  = n0blk + wn * 32 + sn * 16 + (lane & 15);
    float bov = bo[e * DIM + ncol];
#pragma unroll
    for (int sm = 0; sm < 2; ++sm) {
      int mloc = wm * 32 + sm * 16 + ((lane & 16) ? 8 : 0);
#pragma unroll
      for (int r = 0; r < 8; ++r) {
        int t2 = tokS[mloc + r];
        if (t2 >= 0)
          atomicAdd(out + (size_t)t2 * DIM + ncol, cwS[mloc + r] * (acc[sm][sn][r] + bov));
      }
    }
  }
}

// ---------------------------------------------------------------------------
extern "C" void kernel_launch(void* const* d_in, const int* in_sizes, int n_in,
                              void* d_out, int out_size, void* d_ws, size_t ws_size,
                              hipStream_t stream) {
  const float* x      = (const float*)d_in[0];
  const float* gate_w = (const float*)d_in[1];
  const float* wg     = (const float*)d_in[2];
  const float* bg     = (const float*)d_in[3];
  const float* wi     = (const float*)d_in[4];
  const float* bi     = (const float*)d_in[5];
  const float* wo     = (const float*)d_in[6];
  const float* bo     = (const float*)d_in[7];
  float* out = (float*)d_out;

  int N    = in_sizes[0] / DIM;              // 4096 tokens
  int RCAP = 2 * N + NEXP * TM;              // padded routed-row capacity

  char* p = (char*)d_ws;
  auto alloc = [&](size_t bytes) -> char* {
    char* r = p; p += (bytes + 255) & ~(size_t)255; return r;
  };
  int*   counts = (int*)  alloc(NEXP * 4);
  int*   fill   = (int*)  alloc(NEXP * 4);
  int*   base   = (int*)  alloc(NEXP * 4);
  int*   top2e  = (int*)  alloc((size_t)N * 2 * 4);
  float* top2p  = (float*)alloc((size_t)N * 2 * 4);
  int*   tokmap = (int*)  alloc((size_t)RCAP * 4);
  float* cwbuf  = (float*)alloc((size_t)RCAP * 4);
  bf16*  xbf    = (bf16*) alloc((size_t)N * DIM * 2);
  bf16*  hbuf   = (bf16*) alloc((size_t)RCAP * HID * 2);

  int tilesM = (N + TM - 1) / TM;

  moe_init   <<<dim3((out_size + 255) / 256), 256, 0, stream>>>(out, out_size, counts, tokmap, RCAP);
  moe_gate   <<<dim3((N + 7) / 8), 256, 0, stream>>>(x, gate_w, counts, top2e, top2p, N);
  moe_prefix <<<1, 32, 0, stream>>>(counts, base, fill);
  moe_scatter<<<dim3((N + 255) / 256), 256, 0, stream>>>(top2e, top2p, base, fill, tokmap, cwbuf, N);
  moe_cvt_x  <<<dim3((N * DIM / 8 + 255) / 256), 256, 0, stream>>>(x, xbf, N * DIM);
  moe_ffn1   <<<dim3(HID / TN, NEXP * tilesM), 256, 0, stream>>>(xbf, wg, bg, wi, bi, counts, base, tokmap, hbuf, tilesM);
  moe_ffn2   <<<dim3(DIM / TN, NEXP * tilesM), 256, 0, stream>>>(hbuf, wo, bo, counts, base, tokmap, cwbuf, out, tilesM);
}
